// FullSelfAttention_52398601011450
// MI455X (gfx1250) — matrix operand order repaired and checked by reading, not compiled, not run on hardware
//
#include <hip/hip_runtime.h>
#include <hip/hip_bf16.h>

// ---------------- types ----------------
typedef __bf16 bf16;
typedef __attribute__((ext_vector_type(8)))  __bf16 v8bf;
typedef __attribute__((ext_vector_type(16))) __bf16 v16bf;
typedef __attribute__((ext_vector_type(8)))  float  v8f;

#define BDIM   2
#define SDIM   2048
#define DDIM   768
#define HDIM   12
#define DHDIM  64
#define BS     (BDIM*SDIM)          // 4096
#define SCALE_SM 0.125f             // 1/sqrt(64) * smoothing(1.0)
#define GAMMA_C  0.5f

__device__ inline v8f wmma_bf16(v16bf a, v16bf b, v8f c) {
  // D = A(16x32 bf16) * B(32x16 bf16) + C(16x16 f32)
  return __builtin_amdgcn_wmma_f32_16x16x32_bf16(
      /*neg_a=*/false, a, /*neg_b=*/false, b,
      /*c_mod=*/(short)0, c, /*reuse_a=*/false, /*reuse_b=*/false);
}

// ---- CDNA5 async global->LDS copy (guarded; falls back to sync copy) ----
// Builtin signature (per compiler diagnostic): params are pointers to
// <4 x i32> in AS1 (global) and AS3 (LDS), plus const offset and cpol ints.
typedef int v4i_gcc __attribute__((vector_size(16)));
typedef __attribute__((address_space(1))) v4i_gcc as1_v4i;
typedef __attribute__((address_space(3))) v4i_gcc as3_v4i;

#if __has_builtin(__builtin_amdgcn_global_load_async_to_lds_b128)
#define HAVE_ASYNC_LDS 1
#else
#define HAVE_ASYNC_LDS 0
#endif

// ---- CDNA5 Tensor Data Mover (guarded) ----
#if __has_builtin(__builtin_amdgcn_tensor_load_to_lds)
#define HAVE_TDM 1
#else
#define HAVE_TDM 0
#endif
#if __has_include(<hip/amd_detail/amd_gfx1250_TDM.h>)
#define TDM_ARITY6 1
#else
#define TDM_ARITY6 0
#endif

typedef unsigned int gv4u __attribute__((vector_size(16)));
typedef int          gv8i __attribute__((vector_size(32)));
typedef int          gv4i __attribute__((vector_size(16)));

#if HAVE_TDM
// One TDM descriptor: load a 2D tile of 32 x 64 bf16 elements (64B rows,
// 64 rows, row stride 4096 elements) from global into LDS.
__device__ inline void tdm_load_vtile(const bf16* gsrc, bf16* ldst) {
  unsigned long long ga = (unsigned long long)(size_t)gsrc;
  unsigned ldsb = (unsigned)(size_t)ldst;
  // group0: count=1 | lds_addr | global_addr[56:0] | type=2
  gv4u g0 = { 1u, ldsb, (unsigned)ga,
              (unsigned)((ga >> 32) & 0x01FFFFFFu) | (2u << 30) };
  // group1 bit layout (ISA 8.4):
  //  [17:16] data_size=1 (2 bytes); [79:48] tensor_dim0 (large: disable OOB);
  //  [111:80] tensor_dim1=64; [127:112] tile_dim0=32; [143:128] tile_dim1=64;
  //  [207:160] tensor_dim0_stride=4096 elements
  gv8i g1 = {
    (int)(1u << 16),                 // w0: data_size = 2B
    (int)0xFFFF0000,                 // w1: tensor_dim0[15:0]=0xFFFF << 16
    (int)((64u << 16) | 0x7FFFu),    // w2: tensor_dim0[31:16]=0x7FFF, tensor_dim1 lo = 64
    (int)(32u << 16),                // w3: tensor_dim1 hi = 0, tile_dim0 = 32
    (int)64,                         // w4: tile_dim1 = 64, tile_dim2 = 0
    (int)4096,                       // w5: tensor_dim0_stride[31:0] = 4096
    0, 0 };                          // w6,w7: stride hi / dim1 stride = 0
  gv4i gz = { 0, 0, 0, 0 };          // groups 2/3 unused (2D tile)
#if TDM_ARITY6
  gv8i gz8 = { 0, 0, 0, 0, 0, 0, 0, 0 };
  __builtin_amdgcn_tensor_load_to_lds(g0, g1, gz, gz, gz8, 0);
#else
  __builtin_amdgcn_tensor_load_to_lds(g0, g1, gz, gz, 0);
#endif
}
__device__ inline void tensor_wait_le1() {
#if __has_builtin(__builtin_amdgcn_s_wait_tensorcnt)
  __builtin_amdgcn_s_wait_tensorcnt(1);
#else
  asm volatile("s_wait_tensorcnt 1" ::: "memory");
#endif
}
__device__ inline void tensor_wait_0() {
#if __has_builtin(__builtin_amdgcn_s_wait_tensorcnt)
  __builtin_amdgcn_s_wait_tensorcnt(0);
#else
  asm volatile("s_wait_tensorcnt 0" ::: "memory");
#endif
}
#endif  // HAVE_TDM

__device__ inline void async_copy_b128(const bf16* gsrc, bf16* ldst) {
#if HAVE_ASYNC_LDS
  __builtin_amdgcn_global_load_async_to_lds_b128(
      (as1_v4i*)(unsigned long long)(size_t)gsrc,
      (as3_v4i*)(unsigned)(size_t)ldst, 0, 0);
#else
  *(v8bf*)ldst = *(const v8bf*)gsrc;
#endif
}

__device__ inline void async_wait_le1() {
#if HAVE_ASYNC_LDS
#if __has_builtin(__builtin_amdgcn_s_wait_asynccnt)
  __builtin_amdgcn_s_wait_asynccnt(1);
#else
  asm volatile("s_wait_asynccnt 1" ::: "memory");
#endif
#endif
}
__device__ inline void async_wait_0() {
#if HAVE_ASYNC_LDS
#if __has_builtin(__builtin_amdgcn_s_wait_asynccnt)
  __builtin_amdgcn_s_wait_asynccnt(0);
#else
  asm volatile("s_wait_asynccnt 0" ::: "memory");
#endif
#endif
}

// A fragment: row-major A[ld]; contiguous k order: lane half kh holds k0+kh*16 .. +15.
__device__ inline v16bf load_a_frag(const bf16* __restrict__ A, int ld,
                                    int row0, int k0, int lane) {
  int m  = row0 + (lane & 15);
  int kh = lane >> 4;
  const bf16* p = A + (size_t)m * ld + k0 + kh * 16;
  v8bf lo = *(const v8bf*)p;         // K = k0+kh*16 .. +7
  v8bf hi = *(const v8bf*)(p + 8);   // K = k0+kh*16+8 .. +15
  return __builtin_shufflevector(lo, hi, 0,1,2,3,4,5,6,7,8,9,10,11,12,13,14,15);
}

// B fragment when memory holds B^T row-major, i.e. Bt[n][k] (k contiguous):
// b[e] = B[k0 + kh*16 + e][n] -> one contiguous v16bf per lane.
__device__ inline v16bf load_bT_frag(const bf16* __restrict__ Bt, int ld,
                                     int n0, int k0, int lane) {
  int n  = n0 + (lane & 15);
  int kh = lane >> 4;
  return *(const v16bf*)(Bt + (size_t)n * ld + k0 + kh * 16);
}

// ---------------- elementwise helpers ----------------
__global__ void cvt_bf16_kernel(const float* __restrict__ in, bf16* __restrict__ out, int n) {
  int i = blockIdx.x * 256 + threadIdx.x;
  if (i < n) out[i] = (bf16)in[i];
}

// W f32 [K=768][N=768] -> Wt bf16 [N][K]  (so GEMM B-operand loads are contiguous)
__global__ void cvt_transpose_w_kernel(const float* __restrict__ in, bf16* __restrict__ out) {
  int idx = blockIdx.x * 256 + threadIdx.x;
  if (idx >= DDIM * DDIM) return;
  int n = idx / DDIM, k = idx % DDIM;
  out[idx] = (bf16)in[(size_t)k * DDIM + n];
}

// proj f32 [rows][768] (col = d*12 + h)  ->  out bf16 [12][rows][64]
__global__ void split_qk_kernel(const float* __restrict__ in, bf16* __restrict__ out, int rows) {
  size_t idx = (size_t)blockIdx.x * 256 + threadIdx.x;
  size_t per_h = (size_t)rows * DHDIM;
  if (idx >= per_h * HDIM) return;
  int h = (int)(idx / per_h);
  size_t rem = idx - (size_t)h * per_h;
  int r = (int)(rem / DHDIM);
  int d = (int)(rem % DHDIM);
  out[idx] = (bf16)in[(size_t)r * DDIM + d * HDIM + h];
}

// proj f32 [rows][768] -> V transposed bf16 [12][64][rows]
__global__ void split_v_kernel(const float* __restrict__ in, bf16* __restrict__ out, int rows) {
  size_t idx = (size_t)blockIdx.x * 256 + threadIdx.x;
  size_t per_h = (size_t)rows * DHDIM;
  if (idx >= per_h * HDIM) return;
  int h = (int)(idx / per_h);
  size_t rem = idx - (size_t)h * per_h;
  int d = (int)(rem / rows);
  int r = (int)(rem % rows);
  out[idx] = (bf16)in[(size_t)r * DDIM + d * HDIM + h];
}

// ---------------- WMMA GEMM: C(f32)[M][N] = A[M][K] * Bt[N][K]^T ----------------
// block = 128 threads (4 waves), tile 64x64, each wave 32x32 (2x2 WMMA tiles).
__global__ void gemm_bf16t_f32(const bf16* __restrict__ A, const bf16* __restrict__ Bt,
                               float* __restrict__ C, int M, int N, int K) {
  int lane = threadIdx.x & 31;
  int w    = threadIdx.x >> 5;
  int wm = blockIdx.y * 64 + (w & 1) * 32;
  int wn = blockIdx.x * 64 + (w >> 1) * 32;
  v8f acc00 = {}, acc01 = {}, acc10 = {}, acc11 = {};
  for (int k0 = 0; k0 < K; k0 += 32) {
    if (k0 + 32 < K) {
      __builtin_prefetch((const void*)(A  + (size_t)(wm + (lane & 15)) * K + k0 + 32), 0, 3);
      __builtin_prefetch((const void*)(Bt + (size_t)(wn + (lane & 15)) * K + k0 + 32), 0, 3);
    }
    v16bf a0 = load_a_frag (A,  K, wm,      k0, lane);
    v16bf a1 = load_a_frag (A,  K, wm + 16, k0, lane);
    v16bf b0 = load_bT_frag(Bt, K, wn,      k0, lane);
    v16bf b1 = load_bT_frag(Bt, K, wn + 16, k0, lane);
    acc00 = wmma_bf16(a0, b0, acc00);
    acc01 = wmma_bf16(a0, b1, acc01);
    acc10 = wmma_bf16(a1, b0, acc10);
    acc11 = wmma_bf16(a1, b1, acc11);
  }
  int n0 = wn + (lane & 15);
  int mb = wm + (lane >> 4) * 8;
#pragma unroll
  for (int r = 0; r < 8; ++r) {
    C[(size_t)(mb + r)      * N + n0]      = acc00[r];
    C[(size_t)(mb + r)      * N + n0 + 16] = acc01[r];
    C[(size_t)(mb + 16 + r) * N + n0]      = acc10[r];
    C[(size_t)(mb + 16 + r) * N + n0 + 16] = acc11[r];
  }
}

// ---------------- fused scores + softmax (pos stream) ----------------
// Q,Kt: bf16 [12][2048][64]; prob: f32 [12][2048][2048]
// block: 256 threads (8 waves) = 16 query rows x all 2048 keys.
// dynamic LDS: sc[16][2048] f32 + red[16][16] f32
__global__ void attn_pos_kernel(const bf16* __restrict__ Q, const bf16* __restrict__ Kt,
                                float* __restrict__ prob) {
  extern __shared__ float smem[];
  float* sc  = smem;                  // [16][2048]
  float* red = smem + 16 * 2048;      // [16][16]
  int h  = blockIdx.y;
  int i0 = blockIdx.x * 16;
  int lane = threadIdx.x & 31, w = threadIdx.x >> 5;
  const bf16* Qh = Q  + ((size_t)h * SDIM) * DHDIM;
  const bf16* Kh = Kt + ((size_t)h * SDIM) * DHDIM;
  v16bf a0 = load_a_frag(Qh, DHDIM, i0, 0,  lane);
  v16bf a1 = load_a_frag(Qh, DHDIM, i0, 32, lane);
  for (int t = 0; t < 16; ++t) {             // 8 waves x 16 tiles = 128 j-tiles
    int j0 = (w * 16 + t) * 16;
    if (t + 1 < 16)
      __builtin_prefetch((const void*)(Kh + (size_t)(j0 + 16 + (lane & 15)) * DHDIM), 0, 3);
    v16bf b0 = load_bT_frag(Kh, DHDIM, j0, 0,  lane);
    v16bf b1 = load_bT_frag(Kh, DHDIM, j0, 32, lane);
    v8f acc = {};
    acc = wmma_bf16(a0, b0, acc);
    acc = wmma_bf16(a1, b1, acc);
    int n  = j0 + (lane & 15);
    int mb = (lane >> 4) * 8;
#pragma unroll
    for (int r = 0; r < 8; ++r) sc[(size_t)(mb + r) * 2048 + n] = acc[r] * SCALE_SM;
  }
  __syncthreads();
  int row = threadIdx.x >> 4;   // 0..15
  int sub = threadIdx.x & 15;   // 0..15
  float mx = -1e30f;
  for (int j = sub; j < SDIM; j += 16) mx = fmaxf(mx, sc[(size_t)row * 2048 + j]);
  red[row * 16 + sub] = mx;
  __syncthreads();
  mx = -1e30f;
#pragma unroll
  for (int q = 0; q < 16; ++q) mx = fmaxf(mx, red[row * 16 + q]);
  __syncthreads();
  float sum = 0.f;
  for (int j = sub; j < SDIM; j += 16) {
    float e = __expf(sc[(size_t)row * 2048 + j] - mx);
    sc[(size_t)row * 2048 + j] = e;
    sum += e;
  }
  red[row * 16 + sub] = sum;
  __syncthreads();
  sum = 0.f;
#pragma unroll
  for (int q = 0; q < 16; ++q) sum += red[row * 16 + q];
  float inv = 1.0f / sum;
  float* out = prob + ((size_t)h * SDIM + (i0 + row)) * SDIM;
  for (int j = sub; j < SDIM; j += 16)
    __builtin_nontemporal_store(sc[(size_t)row * 2048 + j] * inv, &out[j]);
}

// ---------------- fused scores + softmax + gamma blend (inp stream) ----------------
__global__ void attn_inp_kernel(const bf16* __restrict__ Q, const bf16* __restrict__ Kt,
                                const float* __restrict__ prob_pos,
                                float* __restrict__ attn) {
  extern __shared__ float smem[];
  float* sc  = smem;
  float* red = smem + 16 * 2048;
  int h  = blockIdx.y;
  int b  = blockIdx.z;
  int i0 = blockIdx.x * 16;
  int lane = threadIdx.x & 31, w = threadIdx.x >> 5;
  const bf16* Qh = Q  + ((size_t)h * BS + (size_t)b * SDIM) * DHDIM;
  const bf16* Kh = Kt + ((size_t)h * BS + (size_t)b * SDIM) * DHDIM;
  v16bf a0 = load_a_frag(Qh, DHDIM, i0, 0,  lane);
  v16bf a1 = load_a_frag(Qh, DHDIM, i0, 32, lane);
  for (int t = 0; t < 16; ++t) {
    int j0 = (w * 16 + t) * 16;
    if (t + 1 < 16)
      __builtin_prefetch((const void*)(Kh + (size_t)(j0 + 16 + (lane & 15)) * DHDIM), 0, 3);
    v16bf b0 = load_bT_frag(Kh, DHDIM, j0, 0,  lane);
    v16bf b1 = load_bT_frag(Kh, DHDIM, j0, 32, lane);
    v8f acc = {};
    acc = wmma_bf16(a0, b0, acc);
    acc = wmma_bf16(a1, b1, acc);
    int n  = j0 + (lane & 15);
    int mb = (lane >> 4) * 8;
#pragma unroll
    for (int r = 0; r < 8; ++r) sc[(size_t)(mb + r) * 2048 + n] = acc[r] * SCALE_SM;
  }
  __syncthreads();
  int row = threadIdx.x >> 4;
  int sub = threadIdx.x & 15;
  float mx = -1e30f;
  for (int j = sub; j < SDIM; j += 16) mx = fmaxf(mx, sc[(size_t)row * 2048 + j]);
  red[row * 16 + sub] = mx;
  __syncthreads();
  mx = -1e30f;
#pragma unroll
  for (int q = 0; q < 16; ++q) mx = fmaxf(mx, red[row * 16 + q]);
  __syncthreads();
  float sum = 0.f;
  for (int j = sub; j < SDIM; j += 16) {
    float e = __expf(sc[(size_t)row * 2048 + j] - mx);
    sc[(size_t)row * 2048 + j] = e;
    sum += e;
  }
  red[row * 16 + sub] = sum;
  __syncthreads();
  sum = 0.f;
#pragma unroll
  for (int q = 0; q < 16; ++q) sum += red[row * 16 + q];
  float inv = 1.0f / sum;
  int i = i0 + row;
  const float* pp = prob_pos + ((size_t)h * SDIM + i) * SDIM;
  float* out = attn + (((size_t)b * SDIM + i) * SDIM) * HDIM + h;
  for (int j = sub; j < SDIM; j += 16) {
    float v = (1.0f - GAMMA_C) * (sc[(size_t)row * 2048 + j] * inv) + GAMMA_C * pp[j];
    __builtin_nontemporal_store(v, &out[(size_t)j * HDIM]);
  }
}

// ---------------- PV: out_pre[b][i][h*64+d] = sum_j attn[b,i,j,h] * V[b,j,d,h] ----------------
// attn: f32 [B][S][S][H]; Vt: bf16 [12][64][4096]; outp: bf16 [4096][768]
// block 256 threads (8 waves): 128 i-rows x 64 d-cols. V tile staged in LDS,
// double-buffered, via TDM (preferred) or per-lane async global->LDS copies.
__global__ void pv_kernel(const float* __restrict__ attn, const bf16* __restrict__ Vt,
                          bf16* __restrict__ outp) {
  __shared__ __align__(64) bf16 vtile[2][64 * 32];   // 2 x 4KB
  int h = blockIdx.y, b = blockIdx.z;
  int t = threadIdx.x;
  int lane = t & 31, w = t >> 5;
  int i0 = blockIdx.x * 128 + w * 16;
  const float* A  = attn + ((size_t)b * SDIM) * SDIM * HDIM + h;       // A[i][j] at (i*S + j)*H
  const bf16*  Bv = Vt + ((size_t)h * DHDIM) * BS + (size_t)b * SDIM;  // row d at d*4096 + j
  int crow = t >> 2;
  int cq   = (t & 3) * 8;
  (void)crow; (void)cq;
  const int NSTEP = SDIM / 32;   // 64
#if HAVE_TDM
  if (w == 0) tdm_load_vtile(Bv, &vtile[0][0]);
#else
  async_copy_b128(Bv + (size_t)crow * BS + cq, &vtile[0][crow * 32 + cq]);
#endif
  v8f acc[4] = {};
  for (int js = 0; js < NSTEP; ++js) {
    int j0  = js * 32;
    int cur = js & 1;
#if HAVE_TDM
    if (w == 0) {
      if (js + 1 < NSTEP) {
        tdm_load_vtile(Bv + (j0 + 32), &vtile[cur ^ 1][0]);
        tensor_wait_le1();
      } else {
        tensor_wait_0();
      }
    }
#else
    if (js + 1 < NSTEP) {
      async_copy_b128(Bv + (size_t)crow * BS + (j0 + 32) + cq,
                      &vtile[cur ^ 1][crow * 32 + cq]);
      async_wait_le1();
    } else {
      async_wait_0();
    }
#endif
    __syncthreads();
    // A fragment: strided f32 (h-fastest layout), streamed non-temporally
    int m  = i0 + (lane & 15);
    int kh = lane >> 4;
    const float* ap = A + (size_t)m * SDIM * HDIM;
    v16bf a;
#pragma unroll
    for (int e = 0; e < 16; ++e) {
      int j = j0 + kh * 16 + e;
      a[e] = (bf16)__builtin_nontemporal_load(&ap[(size_t)j * HDIM]);
    }
#pragma unroll
    for (int dt = 0; dt < 4; ++dt) {
      int n = dt * 16 + (lane & 15);
      v16bf bb = *(const v16bf*)(&vtile[cur][n * 32 + kh * 16]);
      acc[dt] = wmma_bf16(a, bb, acc[dt]);
    }
    __syncthreads();
  }
  int mb = i0 + (lane >> 4) * 8;
#pragma unroll
  for (int dt = 0; dt < 4; ++dt) {
    int d = dt * 16 + (lane & 15);
#pragma unroll
    for (int r = 0; r < 8; ++r)
      outp[((size_t)b * SDIM + mb + r) * DDIM + h * DHDIM + d] = (bf16)acc[dt][r];
  }
}

// ---------------- host launcher ----------------
extern "C" void kernel_launch(void* const* d_in, const int* in_sizes, int n_in,
                              void* d_out, int out_size, void* d_ws, size_t ws_size,
                              hipStream_t stream) {
  (void)in_sizes; (void)n_in; (void)out_size; (void)ws_size;
  const float* inp     = (const float*)d_in[0];
  const float* pos_emb = (const float*)d_in[1];
  const float* Ww[6]   = { (const float*)d_in[2], (const float*)d_in[3],
                           (const float*)d_in[4], (const float*)d_in[5],
                           (const float*)d_in[6], (const float*)d_in[7] };
  // Wq_inp, Wk_inp, Wq_pos, Wk_pos, Wv, Wo

  char* ws = (char*)d_ws;
  size_t off = 0;
  auto take = [&](size_t bytes) -> char* {
    char* p = ws + off;
    off += (bytes + 255) & ~(size_t)255;
    return p;
  };

  bf16* inp_b = (bf16*)take((size_t)BS * DDIM * sizeof(bf16));
  bf16* pos_b = (bf16*)take((size_t)SDIM * DDIM * sizeof(bf16));
  bf16* Wt[6];
  for (int i = 0; i < 6; ++i) Wt[i] = (bf16*)take((size_t)DDIM * DDIM * sizeof(bf16));
  float* tmp   = (float*)take((size_t)BS * DDIM * sizeof(float));          // proj scratch
  bf16* Qih = (bf16*)take((size_t)HDIM * BS * DHDIM * sizeof(bf16));
  bf16* Kih = (bf16*)take((size_t)HDIM * BS * DHDIM * sizeof(bf16));
  bf16* Vt  = (bf16*)take((size_t)HDIM * DHDIM * BS * sizeof(bf16));
  bf16* Qph = (bf16*)take((size_t)HDIM * SDIM * DHDIM * sizeof(bf16));
  bf16* Kph = (bf16*)take((size_t)HDIM * SDIM * DHDIM * sizeof(bf16));
  float* prob_pos = (float*)take((size_t)HDIM * SDIM * SDIM * sizeof(float)); // 201 MB
  bf16* outp = (bf16*)take((size_t)BS * DDIM * sizeof(bf16));

  // ---- convert activations to bf16; transpose+convert weights ----
  {
    int n = BS * DDIM;
    cvt_bf16_kernel<<<dim3((n + 255) / 256), dim3(256), 0, stream>>>(inp, inp_b, n);
    n = SDIM * DDIM;
    cvt_bf16_kernel<<<dim3((n + 255) / 256), dim3(256), 0, stream>>>(pos_emb, pos_b, n);
    n = DDIM * DDIM;
    for (int i = 0; i < 6; ++i)
      cvt_transpose_w_kernel<<<dim3((n + 255) / 256), dim3(256), 0, stream>>>(Ww[i], Wt[i]);
  }

  dim3 gemm_blk(128);
  dim3 gemm_grid_inp(DDIM / 64, BS / 64);     // 12 x 64
  dim3 gemm_grid_pos(DDIM / 64, SDIM / 64);   // 12 x 32
  int split_blocks_inp = (int)(((size_t)BS * DDIM + 255) / 256);
  int split_blocks_pos = (int)(((size_t)SDIM * DDIM + 255) / 256);

  // ---- projections + split heads ----
  gemm_bf16t_f32<<<gemm_grid_inp, gemm_blk, 0, stream>>>(inp_b, Wt[0], tmp, BS, DDIM, DDIM);
  split_qk_kernel<<<dim3(split_blocks_inp), dim3(256), 0, stream>>>(tmp, Qih, BS);
  gemm_bf16t_f32<<<gemm_grid_inp, gemm_blk, 0, stream>>>(inp_b, Wt[1], tmp, BS, DDIM, DDIM);
  split_qk_kernel<<<dim3(split_blocks_inp), dim3(256), 0, stream>>>(tmp, Kih, BS);
  gemm_bf16t_f32<<<gemm_grid_inp, gemm_blk, 0, stream>>>(inp_b, Wt[4], tmp, BS, DDIM, DDIM);
  split_v_kernel<<<dim3(split_blocks_inp), dim3(256), 0, stream>>>(tmp, Vt, BS);
  gemm_bf16t_f32<<<gemm_grid_pos, gemm_blk, 0, stream>>>(pos_b, Wt[2], tmp, SDIM, DDIM, DDIM);
  split_qk_kernel<<<dim3(split_blocks_pos), dim3(256), 0, stream>>>(tmp, Qph, SDIM);
  gemm_bf16t_f32<<<gemm_grid_pos, gemm_blk, 0, stream>>>(pos_b, Wt[3], tmp, SDIM, DDIM, DDIM);
  split_qk_kernel<<<dim3(split_blocks_pos), dim3(256), 0, stream>>>(tmp, Kph, SDIM);

  // ---- attention ----
  size_t smem_bytes = (16 * 2048 + 16 * 16) * sizeof(float);  // 129 KB dynamic LDS
  float* attn_out_region  = (float*)d_out;                       // [B,S,D]
  float* attn_prob_region = (float*)d_out + (size_t)BS * DDIM;   // [B,S,S,H]

  attn_pos_kernel<<<dim3(SDIM / 16, HDIM), dim3(256), smem_bytes, stream>>>(
      Qph, Kph, prob_pos);
  attn_inp_kernel<<<dim3(SDIM / 16, HDIM, BDIM), dim3(256), smem_bytes, stream>>>(
      Qih, Kih, prob_pos, attn_prob_region);

  // ---- PV + output projection ----
  pv_kernel<<<dim3(SDIM / 128, HDIM, BDIM), dim3(256), 0, stream>>>(
      attn_prob_region, Vt, outp);
  gemm_bf16t_f32<<<gemm_grid_inp, gemm_blk, 0, stream>>>(outp, Wt[5], attn_out_region,
                                                         BS, DDIM, DDIM);
}
